// DuvenaudMPNN_10179072491921
// MI455X (gfx1250) — matrix operand order repaired and checked by reading, not compiled, not run on hardware
//
#include <hip/hip_runtime.h>
#include <hip/hip_bf16.h>

#define N_NODES 50000
#define N_EDGES 500000
#define F_IN    128
#define F_E     32
#define F_MSG   160
#define N_BK    32
#define T_STEPS 4
#define N_OUT   10
#define MAX_TILES 3200
#define AS 176   // padded A-tile row stride (bf16 elems), 352B = 22*16B

typedef __attribute__((ext_vector_type(16))) __bf16 v16bf;
typedef __attribute__((ext_vector_type(8)))  __bf16 v8bf;
typedef __attribute__((ext_vector_type(8)))  float  v8f;

// ---------------- utility zero kernels ----------------
__global__ void k_zero_f32(float* p, long n) {
    long i = (long)blockIdx.x * blockDim.x + threadIdx.x;
    if (i < n) p[i] = 0.0f;
}
__global__ void k_zero_i32(int* p, int n) {
    int i = blockIdx.x * blockDim.x + threadIdx.x;
    if (i < n) p[i] = 0;
}

// ---------------- one-time weight conversion: f32 [t][b][k][c] -> bf16 [t][b][c][k] ----
__global__ void k_convert_w(const float* __restrict__ lw, __bf16* __restrict__ wb) {
    long i = (long)blockIdx.x * 256 + threadIdx.x;
    const long total = (long)T_STEPS * N_BK * F_IN * F_MSG;
    if (i >= total) return;
    long tb  = i / (F_IN * F_MSG);
    int  rem = (int)(i - tb * (F_IN * F_MSG));
    int  c   = rem / F_MSG;
    int  k   = rem - c * F_MSG;
    wb[i] = (__bf16)lw[tb * (long)(F_MSG * F_IN) + (long)k * F_IN + c];
}

// ---------------- bucket setup (once) ----------------
__global__ void k_bucket(const int* __restrict__ deg, int* __restrict__ bucketOf,
                         float* __restrict__ dinv, int* __restrict__ counts) {
    int i = blockIdx.x * blockDim.x + threadIdx.x;
    if (i >= N_NODES) return;
    int d = deg[i];
    d = d < 1 ? 1 : (d > 32 ? 32 : d);
    bucketOf[i] = d - 1;
    dinv[i] = 1.0f / (float)d;
    atomicAdd(&counts[d - 1], 1);
}

__global__ void k_scan(const int* __restrict__ counts, int* __restrict__ offsets,
                       int* __restrict__ tileBucket, int* __restrict__ tileBase,
                       int* __restrict__ tileLimit, int* __restrict__ meta) {
    if (threadIdx.x != 0 || blockIdx.x != 0) return;
    int off = 0, tt = 0;
    for (int b = 0; b < N_BK; ++b) {
        offsets[b] = off;
        int c  = counts[b];
        int nt = (c + 15) >> 4;
        for (int i = 0; i < nt; ++i) {
            tileBucket[tt] = b;
            tileBase[tt]   = off + i * 16;
            tileLimit[tt]  = off + c;
            ++tt;
        }
        off += c;
    }
    meta[0] = tt;
}

__global__ void k_scatter(const int* __restrict__ bucketOf, const int* __restrict__ offsets,
                          int* __restrict__ cursors, int* __restrict__ perm) {
    int i = blockIdx.x * blockDim.x + threadIdx.x;
    if (i >= N_NODES) return;
    int b = bucketOf[i];
    int pos = offsets[b] + atomicAdd(&cursors[b], 1);
    perm[pos] = i;
}

// ---------------- phase 1: gather + scatter-add aggregation ----------------
__global__ void __launch_bounds__(256) k_aggregate(const float* __restrict__ z,
                                                   const float* __restrict__ eattr,
                                                   const int*   __restrict__ eidx,
                                                   float* __restrict__ agg) {
    int e = blockIdx.x * 8 + (threadIdx.x >> 5);
    if (e >= N_EDGES) return;
    int lane = threadIdx.x & 31;
    int src = eidx[e];
    int dst = eidx[N_EDGES + e];
    const float* zs = z + (long)src * F_IN;
    float* ad = agg + (long)dst * F_MSG;
#pragma unroll
    for (int it = 0; it < 5; ++it) {
        int f = lane + it * 32;
        float v = (it < 4) ? zs[f] : eattr[(long)e * F_E + (f - F_IN)];
        unsafeAtomicAdd(&ad[f], v);
    }
}

// ---------------- phase 2: per-bucket WMMA GEMM + sigmoid ----------------
// block = 256 thr = 8 waves; wave w covers output cols [16w, 16w+16)
// C[16x128] = (agg[rows]/d -> bf16)[16x160] @ Wb(bf16)[160x128], K split as 5x32
__global__ void __launch_bounds__(256) k_bucket_gemm(
        const float* __restrict__ agg, const float* __restrict__ dinv,
        const int* __restrict__ perm,
        const int* __restrict__ tileBucket, const int* __restrict__ tileBase,
        const int* __restrict__ tileLimit,  const int* __restrict__ meta,
        const __bf16* __restrict__ Wbf_t,   // [32][128][160] bf16 (col-major-K)
        float* __restrict__ zout) {
    __shared__ __align__(16) __bf16 Wl[F_IN * F_MSG]; // Wl[col*160 + k], 40 KB
    __shared__ __align__(16) __bf16 Al[16 * AS];      // Al[row*176 + k]

    int tt = blockIdx.x;
    if (tt >= meta[0]) return;
    int b = tileBucket[tt], base = tileBase[tt], limit = tileLimit[tt];
    int tid = threadIdx.x;

    // stage bucket weight: pure b128 copy, layout already matches LDS layout
    const uint4* wsrc = (const uint4*)(Wbf_t + (size_t)b * F_IN * F_MSG);
    uint4* wdst = (uint4*)Wl;
#pragma unroll
    for (int i = 0; i < 10; ++i) wdst[tid + i * 256] = wsrc[tid + i * 256];

    // stage A tile: gather 16 permuted node rows of agg, scale by 1/d, cvt bf16
#pragma unroll
    for (int i = 0; i < 10; ++i) {
        int idx = tid + i * 256;
        int r = idx / F_MSG, f = idx - r * F_MSG;
        int gr = base + r;
        float v = 0.0f;
        if (gr < limit) {
            int node = perm[gr];
            v = agg[(long)node * F_MSG + f] * dinv[node];
        }
        Al[r * AS + f] = (__bf16)v;
    }
    __syncthreads();

    int lane = tid & 31, wave = tid >> 5;
    int lo = lane & 15, hi = lane >> 4;
    int col = wave * 16 + lo;

    v8f acc = {};
#pragma unroll
    for (int kk = 0; kk < 5; ++kk) {
        int kbase = kk * 32;
        // A fragment (16x32 bf16): row=lo, elem j -> K=kbase+16*hi+j
        v8bf a0 = *(const v8bf*)&Al[lo * AS + kbase + hi * 16];
        v8bf a1 = *(const v8bf*)&Al[lo * AS + kbase + hi * 16 + 8];
        v16bf afr = __builtin_shufflevector(a0, a1, 0,1,2,3,4,5,6,7,8,9,10,11,12,13,14,15);
        // B fragment (32x16 bf16): col=lo, elem j -> K=kbase+16*hi+j (contiguous in Wl)
        const v8bf* bp = (const v8bf*)&Wl[col * F_MSG + kbase + hi * 16];
        v16bf bfr = __builtin_shufflevector(bp[0], bp[1], 0,1,2,3,4,5,6,7,8,9,10,11,12,13,14,15);
        acc = __builtin_amdgcn_wmma_f32_16x16x32_bf16(
                  false, afr, false, bfr, (short)0, acc, false, false);
    }

    // C layout: vgpr r, lanes 0-15: M=r; lanes 16-31: M=r+8; N=lo
#pragma unroll
    for (int r = 0; r < 8; ++r) {
        int M = r + hi * 8;
        int gr = base + M;
        if (gr < limit) {
            int node = perm[gr];
            float v = acc[r];
            // sigmoid with hw rcp (v_rcp_f32) instead of IEEE divide
            zout[(long)node * F_IN + col] =
                __builtin_amdgcn_rcpf(1.0f + __expf(-v));
        }
    }
}

// ---------------- phase 3: readout (z @ R, softmax, global sum) ----------------
__global__ void __launch_bounds__(256) k_readout(const float* __restrict__ z,
                                                 const float* __restrict__ R,
                                                 float* __restrict__ out) {
    __shared__ float Rl[F_IN * N_OUT]; // 5 KB
    __shared__ float sAcc[N_OUT];
    int tid = threadIdx.x;
    for (int i = tid; i < F_IN * N_OUT; i += 256) Rl[i] = R[i];
    if (tid < N_OUT) sAcc[tid] = 0.0f;
    __syncthreads();

    int wave = tid >> 5, lane = tid & 31;
    int node = blockIdx.x * 8 + wave;
    if (node < N_NODES) {
        const float4* row = (const float4*)(z + (long)node * F_IN);
        float4 v = row[lane];
        float logit[N_OUT];
#pragma unroll
        for (int o = 0; o < N_OUT; ++o) {
            int f = lane * 4;
            float a = v.x * Rl[(f + 0) * N_OUT + o] + v.y * Rl[(f + 1) * N_OUT + o]
                    + v.z * Rl[(f + 2) * N_OUT + o] + v.w * Rl[(f + 3) * N_OUT + o];
#pragma unroll
            for (int off = 16; off > 0; off >>= 1) a += __shfl_down(a, off, 32);
            logit[o] = a; // valid on lane 0
        }
        if (lane == 0) {
            float m = logit[0];
#pragma unroll
            for (int o = 1; o < N_OUT; ++o) m = fmaxf(m, logit[o]);
            float s = 0.0f, e[N_OUT];
#pragma unroll
            for (int o = 0; o < N_OUT; ++o) { e[o] = __expf(logit[o] - m); s += e[o]; }
            float inv = __builtin_amdgcn_rcpf(s);
#pragma unroll
            for (int o = 0; o < N_OUT; ++o) atomicAdd(&sAcc[o], e[o] * inv);
        }
    }
    __syncthreads();
    if (tid < N_OUT) unsafeAtomicAdd(&out[tid], sAcc[tid]);
}

// ---------------- host orchestration ----------------
extern "C" void kernel_launch(void* const* d_in, const int* in_sizes, int n_in,
                              void* d_out, int out_size, void* d_ws, size_t ws_size,
                              hipStream_t stream) {
    const float* x     = (const float*)d_in[0];
    const float* eattr = (const float*)d_in[1];
    const float* lw    = (const float*)d_in[2]; // [T][32][160][128]
    const float* rw    = (const float*)d_in[3]; // [T][128][10]
    const int*   eidx  = (const int*)d_in[4];   // [2][E]
    const int*   deg   = (const int*)d_in[5];
    float* out = (float*)d_out;

    char* ws = (char*)d_ws;
    const size_t OFF_AGG  = 0;
    const size_t OFF_ZA   = OFF_AGG + (size_t)N_NODES * F_MSG * 4;  // 32 MB
    const size_t OFF_ZB   = OFF_ZA  + (size_t)N_NODES * F_IN  * 4;  // +25.6 MB
    const size_t OFF_DINV = OFF_ZB  + (size_t)N_NODES * F_IN  * 4;  // +25.6 MB
    const size_t OFF_BKT  = OFF_DINV + (size_t)N_NODES * 4;
    const size_t OFF_PERM = OFF_BKT  + (size_t)N_NODES * 4;
    const size_t OFF_CNT  = OFF_PERM + (size_t)N_NODES * 4;
    const size_t OFF_OFFS = OFF_CNT  + 128;
    const size_t OFF_CUR  = OFF_OFFS + 128;
    const size_t OFF_META = OFF_CUR  + 128;
    const size_t OFF_TB   = OFF_META + 64;
    const size_t OFF_TBASE= OFF_TB   + (size_t)MAX_TILES * 4;
    const size_t OFF_TLIM = OFF_TBASE+ (size_t)MAX_TILES * 4;
    const size_t OFF_WBF  = OFF_TLIM + (size_t)MAX_TILES * 4;       // bf16 weights, ~5.2 MB

    float* agg  = (float*)(ws + OFF_AGG);
    float* zA   = (float*)(ws + OFF_ZA);
    float* zB   = (float*)(ws + OFF_ZB);
    float* dinv = (float*)(ws + OFF_DINV);
    int* bucketOf = (int*)(ws + OFF_BKT);
    int* perm     = (int*)(ws + OFF_PERM);
    int* counts   = (int*)(ws + OFF_CNT);
    int* offsets  = (int*)(ws + OFF_OFFS);
    int* cursors  = (int*)(ws + OFF_CUR);
    int* meta     = (int*)(ws + OFF_META);
    int* tileBucket = (int*)(ws + OFF_TB);
    int* tileBase   = (int*)(ws + OFF_TBASE);
    int* tileLimit  = (int*)(ws + OFF_TLIM);
    __bf16* wbf     = (__bf16*)(ws + OFF_WBF);

    // init: zero output accumulator + bucket counters
    k_zero_f32<<<1, 32, 0, stream>>>(out, N_OUT);
    k_zero_i32<<<1, 64, 0, stream>>>(counts, N_BK);
    k_zero_i32<<<1, 64, 0, stream>>>(cursors, N_BK);

    // one-time bf16 transposed weight cache
    const long wTotal = (long)T_STEPS * N_BK * F_IN * F_MSG;
    k_convert_w<<<(int)((wTotal + 255) / 256), 256, 0, stream>>>(lw, wbf);

    // degree-bucket partition (static across timesteps)
    int nblk = (N_NODES + 255) / 256;
    k_bucket <<<nblk, 256, 0, stream>>>(deg, bucketOf, dinv, counts);
    k_scan   <<<1, 64, 0, stream>>>(counts, offsets, tileBucket, tileBase, tileLimit, meta);
    k_scatter<<<nblk, 256, 0, stream>>>(bucketOf, offsets, cursors, perm);

    const long aggN = (long)N_NODES * F_MSG;
    const int aggBlk = (int)((aggN + 255) / 256);

    for (int t = 0; t < T_STEPS; ++t) {
        const float* zin  = (t == 0) ? x : ((t & 1) ? zA : zB);
        float*       zout = (t & 1) ? zB : zA;
        const __bf16* Wt = wbf + (size_t)t * N_BK * F_IN * F_MSG;
        const float*  Rt = rw  + (size_t)t * F_IN * N_OUT;

        k_zero_f32<<<aggBlk, 256, 0, stream>>>(agg, aggN);
        k_aggregate<<<(N_EDGES + 7) / 8, 256, 0, stream>>>(zin, eattr, eidx, agg);
        k_bucket_gemm<<<MAX_TILES, 256, 0, stream>>>(agg, dinv, perm, tileBucket,
                                                     tileBase, tileLimit, meta, Wt, zout);
        k_readout<<<(N_NODES + 7) / 8, 256, 0, stream>>>(zout, Rt, out);
    }
}
